// DeformableConv2d_52304111731231
// MI455X (gfx1250) — matrix operand order repaired and checked by reading, not compiled, not run on hardware
//
#include <hip/hip_runtime.h>
#include <hip/hip_bf16.h>
#include <math.h>

typedef __attribute__((ext_vector_type(16))) _Float16 v16h;
typedef __attribute__((ext_vector_type(8)))  float    v8f;

#define Bn   4
#define Cc   64
#define Hh   128
#define Ww   128
#define HW   (Hh*Ww)          // 16384
#define CHW  (Cc*Hh*Ww)
#define Oo   64
#define KK9  9
#define Kdim 576              // Cc * KK9
#define NSTEP 18              // Kdim / 32

// ---- workspace layout (bytes) ----
#define OFF_OFFSET 0
#define OFF_SIZE   (Bn*18*Hh*Ww*4)             // 4,718,592
#define MSK_OFFSET (OFF_OFFSET + OFF_SIZE)
#define MSK_SIZE   (Bn*9*Hh*Ww*4)              // 2,359,296
#define WP1_OFFSET (MSK_OFFSET + MSK_SIZE)     // 7,077,888 (32B aligned)
#define WP1_ELTS   (NSTEP*2*32*16)             // 18432 halves
#define WP2_OFFSET (WP1_OFFSET + WP1_ELTS*2)   // 7,114,752 (32B aligned)
#define WP2_ELTS   (NSTEP*4*32*16)             // 36864 halves

// K ordering for the GEMM is kk-major: k_gemm = kk*64 + c.
// A-fragment k order chosen by this code for v_wmma_f32_16x16x32_f16 (16-bit A, 16x32):
// lanes 0-15: K=j ; lanes 16-31: K=j+16 (contiguous, matches B fills)
__device__ __forceinline__ int a_klocal(int lane, int j) {
    return (lane < 16) ? j : j + 16;
}

// ---------------- weight packing (A fragments, kk-major K) ----------------
__global__ __launch_bounds__(256) void prep_weights(
    const float* __restrict__ w_off, const float* __restrict__ w_mod,
    const float* __restrict__ w_reg, _Float16* __restrict__ wp1,
    _Float16* __restrict__ wp2) {
  int i = blockIdx.x * 256 + threadIdx.x;
  if (i < WP1_ELTS) {
    int j = i & 15, lane = (i >> 4) & 31, mt = (i >> 9) & 1, ks = i >> 10;
    int m = mt * 16 + (lane & 15);
    int k = ks * 32 + a_klocal(lane, j);
    int c = k & 63, kk = k >> 6;                 // kk-major ordering
    float w = 0.f;
    if (m < 18)       w = w_off[(m * Cc + c) * KK9 + kk];
    else if (m < 27)  w = w_mod[((m - 18) * Cc + c) * KK9 + kk];
    wp1[i] = (_Float16)w;
  } else {
    int t = i - WP1_ELTS;
    if (t >= WP2_ELTS) return;
    int j = t & 15, lane = (t >> 4) & 31, mt = (t >> 9) & 3, ks = t >> 11;
    int m = mt * 16 + (lane & 15);
    int k = ks * 32 + a_klocal(lane, j);
    int c = k & 63, kk = k >> 6;
    wp2[t] = (_Float16)w_reg[(m * Cc + c) * KK9 + kk];
  }
}

// ---------------- offset/mask conv (WMMA GEMM, M=32pad, K=576, N=16) ----------------
__global__ __launch_bounds__(256) void conv_offmask(
    const float* __restrict__ x, const float* __restrict__ b_off,
    const float* __restrict__ b_mod, const _Float16* __restrict__ wp1,
    float* __restrict__ off, float* __restrict__ msk) {
  const int lane = threadIdx.x & 31;
  const int wv   = threadIdx.x >> 5;
  const int tile = blockIdx.x * 8 + wv;
  const int base = tile * 16;
  const int b    = base >> 14;          // / (H*W)
  const int rem  = base & 16383;
  const int y    = rem >> 7;            // / W
  const int x0   = rem & 127;
  const int p    = lane & 15;
  const int xp   = x0 + p;
  // lanes 16-31 handle channels c+16 within each 32-wide K half
  const float* xlane = x + b * CHW + ((lane < 16) ? 0 : 16) * HW;
  const v16h* wpv = (const v16h*)wp1;

  v8f acc0 = {}; v8f acc1 = {};
  for (int kk = 0; kk < KK9; ++kk) {
    int ky = kk / 3, kx = kk - 3 * (kk / 3);     // once per kk, not per element
    int yy = y + ky - 1, xx = xp + kx - 1;
    float validf = (yy >= 0 && yy < Hh && xx >= 0 && xx < Ww) ? 1.f : 0.f;
    int yyc = yy < 0 ? 0 : (yy > Hh - 1 ? Hh - 1 : yy);
    int xxc = xx < 0 ? 0 : (xx > Ww - 1 ? Ww - 1 : xx);
    const float* src = xlane + yyc * Ww + xxc;   // branchless: clamp + 0/1 multiply
    const v16h* wq = wpv + kk * 128 + lane;      // ((kk*2+half)*2+mt)*32 + lane
#pragma unroll
    for (int half = 0; half < 2; ++half) {
      v16h bf;
#pragma unroll
      for (int j = 0; j < 16; ++j)               // immediate-offset channel gathers
        bf[j] = (_Float16)(validf * src[(half * 32 + j) * HW]);
      v16h a0 = wq[(half * 2 + 0) * 32];
      v16h a1 = wq[(half * 2 + 1) * 32];
      acc0 = __builtin_amdgcn_wmma_f32_16x16x32_f16(false, a0, false, bf, (short)0, acc0, false, false);
      acc1 = __builtin_amdgcn_wmma_f32_16x16x32_f16(false, a1, false, bf, (short)0, acc1, false, false);
    }
  }
  const int rowadd = (lane < 16) ? 0 : 8;
#pragma unroll
  for (int j = 0; j < 8; ++j) {
    int ch0 = j + rowadd;                                   // 0..15 -> offsets
    off[((b * 18 + ch0) * Hh + y) * Ww + xp] = acc0[j] + b_off[ch0];
    int ch1 = 16 + j + rowadd;                              // 16..31
    if (ch1 < 18) {
      off[((b * 18 + ch1) * Hh + y) * Ww + xp] = acc1[j] + b_off[ch1];
    } else if (ch1 < 27) {
      int mc = ch1 - 18;
      float v = acc1[j] + b_mod[mc];
      msk[((b * 9 + mc) * Hh + y) * Ww + xp] = 2.f / (1.f + __expf(-v));
    }
  }
}

// ---------------- deformable GEMM (M=64, K=576, N=16 per wave) ----------------
__global__ __launch_bounds__(256) void deform_gemm(
    const float* __restrict__ x, const float* __restrict__ off,
    const float* __restrict__ msk, const _Float16* __restrict__ wp2,
    float* __restrict__ out) {
  __shared__ int4   sidx[8][144];
  __shared__ float4 swgt[8][144];
  const int lane = threadIdx.x & 31;
  const int wv   = threadIdx.x >> 5;
  const int tile = blockIdx.x * 8 + wv;
  const int base = tile * 16;
  const int b    = base >> 14;
  const int rem  = base & 16383;
  const int y    = rem >> 7;
  const int x0   = rem & 127;

  // ---- bilinear metadata: 16 pixels x 9 taps per wave ----
  for (int t = lane; t < 144; t += 32) {
    int kk = t >> 4, p = t & 15, xp = x0 + p;
    int ky = kk / 3, kx = kk - 3 * (kk / 3);
    float dy = off[((b * 18 + 2 * kk    ) * Hh + y) * Ww + xp];
    float dx = off[((b * 18 + 2 * kk + 1) * Hh + y) * Ww + xp];
    float m  = msk[((b * 9 + kk) * Hh + y) * Ww + xp];
    float py = dy + (float)(y - 1 + ky);
    float px = dx + (float)(xp - 1 + kx);
    float y0f = floorf(py), x0f = floorf(px);
    float wy = py - y0f, wx = px - x0f;
    int iy0 = (int)y0f, ix0 = (int)x0f;
    float wts[4] = { (1.f - wy) * (1.f - wx) * m, (1.f - wy) * wx * m,
                     wy * (1.f - wx) * m,         wy * wx * m };
    int   idx[4];
    float wgt[4];
#pragma unroll
    for (int q = 0; q < 4; ++q) {
      int iy = iy0 + (q >> 1), ix = ix0 + (q & 1);
      bool valid = (iy >= 0) && (iy < Hh) && (ix >= 0) && (ix < Ww);
      int cy = iy < 0 ? 0 : (iy > Hh - 1 ? Hh - 1 : iy);
      int cx = ix < 0 ? 0 : (ix > Ww - 1 ? Ww - 1 : ix);
      idx[q] = cy * Ww + cx;
      wgt[q] = valid ? wts[q] : 0.f;
    }
    sidx[wv][t] = make_int4(idx[0], idx[1], idx[2], idx[3]);
    swgt[wv][t] = make_float4(wgt[0], wgt[1], wgt[2], wgt[3]);
  }
  __syncthreads();

  const float* xlane = x + b * CHW + ((lane < 16) ? 0 : 16) * HW;
  const v16h* wpv = (const v16h*)wp2;
  v8f acc0 = {}, acc1 = {}, acc2 = {}, acc3 = {};
  const int p = lane & 15;
  for (int kk = 0; kk < KK9; ++kk) {
    int4   ii = sidx[wv][kk * 16 + p];           // one b128 LDS read per kk
    float4 ww = swgt[wv][kk * 16 + p];
    const float* s0 = xlane + ii.x;              // 4 corner base pointers per kk
    const float* s1 = xlane + ii.y;
    const float* s2 = xlane + ii.z;
    const float* s3 = xlane + ii.w;
    const v16h* wq = wpv + kk * 256 + lane;      // ((kk*2+half)*4+mt)*32 + lane
#pragma unroll
    for (int half = 0; half < 2; ++half) {
      v16h bf;
#pragma unroll
      for (int j = 0; j < 16; ++j) {             // immediate-offset channel gathers
        int o = (half * 32 + j) * HW;
        bf[j] = (_Float16)(ww.x * s0[o] + ww.y * s1[o] + ww.z * s2[o] + ww.w * s3[o]);
      }
      v16h a0 = wq[(half * 4 + 0) * 32];
      v16h a1 = wq[(half * 4 + 1) * 32];
      v16h a2 = wq[(half * 4 + 2) * 32];
      v16h a3 = wq[(half * 4 + 3) * 32];
      acc0 = __builtin_amdgcn_wmma_f32_16x16x32_f16(false, a0, false, bf, (short)0, acc0, false, false);
      acc1 = __builtin_amdgcn_wmma_f32_16x16x32_f16(false, a1, false, bf, (short)0, acc1, false, false);
      acc2 = __builtin_amdgcn_wmma_f32_16x16x32_f16(false, a2, false, bf, (short)0, acc2, false, false);
      acc3 = __builtin_amdgcn_wmma_f32_16x16x32_f16(false, a3, false, bf, (short)0, acc3, false, false);
    }
  }
  const int rowadd = (lane < 16) ? 0 : 8;
  const int xp = x0 + p;
#pragma unroll
  for (int j = 0; j < 8; ++j) {
    int o = j + rowadd;
    out[((b * Oo + o     ) * Hh + y) * Ww + xp] = acc0[j];
    out[((b * Oo + o + 16) * Hh + y) * Ww + xp] = acc1[j];
    out[((b * Oo + o + 32) * Hh + y) * Ww + xp] = acc2[j];
    out[((b * Oo + o + 48) * Hh + y) * Ww + xp] = acc3[j];
  }
}

extern "C" void kernel_launch(void* const* d_in, const int* in_sizes, int n_in,
                              void* d_out, int out_size, void* d_ws, size_t ws_size,
                              hipStream_t stream) {
  const float* x     = (const float*)d_in[0];
  const float* w_off = (const float*)d_in[1];
  const float* b_off = (const float*)d_in[2];
  const float* w_mod = (const float*)d_in[3];
  const float* b_mod = (const float*)d_in[4];
  const float* w_reg = (const float*)d_in[5];
  char* ws = (char*)d_ws;
  float*    off = (float*)(ws + OFF_OFFSET);
  float*    msk = (float*)(ws + MSK_OFFSET);
  _Float16* wp1 = (_Float16*)(ws + WP1_OFFSET);
  _Float16* wp2 = (_Float16*)(ws + WP2_OFFSET);
  float* out = (float*)d_out;

  prep_weights<<<(WP1_ELTS + WP2_ELTS + 255) / 256, 256, 0, stream>>>(w_off, w_mod, w_reg, wp1, wp2);
  conv_offmask<<<2048, 256, 0, stream>>>(x, b_off, b_mod, wp1, off, msk);
  deform_gemm<<<2048, 256, 0, stream>>>(x, off, msk, wp2, out);
}
